// NonLocalBlock2D_49478023250259
// MI455X (gfx1250) — matrix operand order repaired and checked by reading, not compiled, not run on hardware
//
#include <hip/hip_runtime.h>

// NonLocalBlock2D for MI455X (gfx1250): bf16 WMMA pipeline, async-LDS staging.
//
// Layouts chosen so every GEMM operand is contiguous in K:
//   xbfT [B*N][C]   (x transposed+converted, LDS-tiled transpose)
//   P    [512][B*N] rows 0..255 = g, rows 256..511 = phi
//   thT  [B*N][I]   theta stored transposed by the projection epilogue
//   Mt   [B][I][I]  Mt[i][j] = (ph g^T)[j][i] / N
//   ybfT [B*N][I]   y stored transposed by the y epilogue
//
// GEMM: 4 wave32 / 128-thread block, 128x64 tile, wave tile 32x64, K-step 32,
// LDS double buffered. Staging = 6x global_load_async_to_lds_b128 per thread
// (fragment-order swizzle folded into LDS dest addresses), s_wait_asynccnt 6
// steady state; per wave per step: 12x ds_load_b128 + 8x v_wmma. Last K-step
// peeled so the steady-state loop is branch-free.

typedef unsigned short u16;
typedef unsigned int   u32;

typedef __attribute__((ext_vector_type(16))) __bf16 v16bf;
typedef __attribute__((ext_vector_type(8)))  float  v8f;

#define TILE_M 128
#define TILE_N 64
#define TILE_K 32

#define CC 512            // channels
#define II 256            // inner channels
#define BB 16             // batch
#define NN 3136           // H*W
#define NTOT (BB * NN)    // 50176 = 784*64
#define M3 (3 * II)       // 768

// ---------------- helpers ----------------

__device__ __forceinline__ u16 f2bf(float f) {
  union { float f; u32 u; } v; v.f = f;
  u32 r = v.u + 0x7FFFu + ((v.u >> 16) & 1u);   // round-to-nearest-even
  return (u16)(r >> 16);
}

__device__ __forceinline__ v16bf ldfrag(const u16* s, int lane) {
  // one lane's fragment = 16 contiguous bf16 (two ds_load_b128)
  union { uint4 q[2]; v16bf v; } u;
  const uint4* p = (const uint4*)(s + lane * 16);
  u.q[0] = p[0];
  u.q[1] = p[1];
  return u.v;
}

// async 16B global -> 16B LDS copy (CDNA5 GLOBAL_LOAD_ASYNC_TO_LDS_B128).
__device__ __forceinline__ void async_b128(u32 ldsOff, const u16* g) {
  asm volatile("global_load_async_to_lds_b128 %0, %1, off"
               :: "v"(ldsOff), "v"(g) : "memory");
}
#define WAIT_ASYNC_6() asm volatile("s_wait_asynccnt 0x6" ::: "memory")
#define WAIT_ASYNC_0() asm volatile("s_wait_asynccnt 0x0" ::: "memory")

// ---------------- GEMM ----------------
// All B operands are B^T row-major (rows = output columns, contiguous in K).
// EPI: 0 = +bias; rows [0,I) -> Out(g), [I,2I) -> aux transposed (thT),
//          [2I,3I) -> Out rows m-I (phi)                       [projection]
//      1 = *scale, store transposed bf16 (Out[col][m])         [M and y]
//      3 = +bias +residual, store fp32 [B,C,H,W]               [final]
// ABDIAG: A block selected per 3136-column batch block (y-GEMM).

template<int EPI, bool ABDIAG>
__global__ __launch_bounds__(128) void gemm_bf16_kernel(
    const u16* __restrict__ A, int lda, long long aBOff,
    const u16* __restrict__ B, int ldb, long long bBOff,
    void* __restrict__ Out, int ldo, long long oBOff,
    const float* __restrict__ bias,
    const float* __restrict__ resid,
    u16* __restrict__ aux,
    float scale,
    int K)
{
  __shared__ __align__(16) u16 sA[2 * 8 * 512];   // double-buffered, frag order
  __shared__ __align__(16) u16 sB[2 * 4 * 512];

  const int tid  = threadIdx.x;                   // 0..127
  const int lane = tid & 31;
  const int wave = tid >> 5;                      // 4 waves, 32 m-rows each

  const int       mBase = blockIdx.y * TILE_M;
  const long long nBase = (long long)blockIdx.x * TILE_N;
  const int       z     = blockIdx.z;

  const u16* Ap = A + (ABDIAG ? (long long)(nBase / NN) * aBOff
                              : (long long)z * aBOff);
  const u16* Bp = B + (long long)z * bBOff;

  v8f acc[2][4];
  #pragma unroll
  for (int i = 0; i < 2; i++)
    #pragma unroll
    for (int j = 0; j < 4; j++) acc[i][j] = (v8f)0.0f;

  // ---- per-thread staging (all 16B chunks, contiguous LDS dests) ----
  // A: thread t owns row t of the 128x32 tile: 4 chunks (64B contiguous).
  //    A fragment: lane = (k%16>=8)*16 + m%16, elem = (k/16)*8 + k%8
  const int a_t  = tid >> 4;            // m-subtile
  const int a_mr = tid & 15;
  const u16* rowA = Ap + (size_t)(mBase + tid) * (size_t)lda;
  u32 dA[4];
  const u32 baseA = (u32)(size_t)&sA[0];
  #pragma unroll
  for (int k8 = 0; k8 < 4; k8++)
    dA[k8] = baseA + (u32)(a_t * 512 + ((k8 & 1) * 16 + a_mr) * 16
                           + (k8 >> 1) * 8) * 2;
  // B^T: thread t owns k 8h..8h+7 and 16+8h..16+8h+7 of row (t>>1) (h = t&1),
  //      landing in one contiguous 32B lane slot of the fragment layout.
  //    B fragment: lane = (k%16>=8)*16 + n%16, elem = (k/16)*8 + k%8
  const int bt_i  = tid >> 1;           // 0..63 output column
  const int bt_kh = tid & 1;            // lane half (k 8h..8h+7, 16+8h..16+8h+7)
  const u16* rowB = Bp + (size_t)(nBase + bt_i) * (size_t)ldb + bt_kh * 8;
  const u32 baseB = (u32)(size_t)&sB[0];
  const u32 dB = baseB + (u32)((bt_i >> 4) * 512
                               + (bt_kh * 16 + (bt_i & 15)) * 16) * 2;

  #define ISSUE(kb, buf)                                                    \
    do {                                                                    \
      async_b128(dA[0] + (buf) * 16384, rowA + (kb));                       \
      async_b128(dA[1] + (buf) * 16384, rowA + (kb) + 8);                   \
      async_b128(dA[2] + (buf) * 16384, rowA + (kb) + 16);                  \
      async_b128(dA[3] + (buf) * 16384, rowA + (kb) + 24);                  \
      async_b128(dB     + (buf) * 8192, rowB + (kb));                       \
      async_b128(dB + 16 + (buf) * 8192, rowB + (kb) + 16);                 \
    } while (0)

  auto compute = [&](int buf) {
    const u16* cA = &sA[buf * 8192];
    const u16* cB = &sB[buf * 4096];
    v16bf a0 = ldfrag(cA + (wave * 2 + 0) * 512, lane);
    v16bf a1 = ldfrag(cA + (wave * 2 + 1) * 512, lane);
    v16bf b0 = ldfrag(cB + 0 * 512, lane);
    v16bf b1 = ldfrag(cB + 1 * 512, lane);
    v16bf b2 = ldfrag(cB + 2 * 512, lane);
    v16bf b3 = ldfrag(cB + 3 * 512, lane);
    acc[0][0] = __builtin_amdgcn_wmma_f32_16x16x32_bf16(false, a0, false, b0, (short)0, acc[0][0], false, false);
    acc[0][1] = __builtin_amdgcn_wmma_f32_16x16x32_bf16(false, a0, false, b1, (short)0, acc[0][1], false, false);
    acc[0][2] = __builtin_amdgcn_wmma_f32_16x16x32_bf16(false, a0, false, b2, (short)0, acc[0][2], false, false);
    acc[0][3] = __builtin_amdgcn_wmma_f32_16x16x32_bf16(false, a0, false, b3, (short)0, acc[0][3], false, false);
    acc[1][0] = __builtin_amdgcn_wmma_f32_16x16x32_bf16(false, a1, false, b0, (short)0, acc[1][0], false, false);
    acc[1][1] = __builtin_amdgcn_wmma_f32_16x16x32_bf16(false, a1, false, b1, (short)0, acc[1][1], false, false);
    acc[1][2] = __builtin_amdgcn_wmma_f32_16x16x32_bf16(false, a1, false, b2, (short)0, acc[1][2], false, false);
    acc[1][3] = __builtin_amdgcn_wmma_f32_16x16x32_bf16(false, a1, false, b3, (short)0, acc[1][3], false, false);
  };

  ISSUE(0, 0);
  int ibuf = 0;
  int kb = 0;
  // steady state: branch-free (last step peeled)
  for (; kb < K - TILE_K; kb += TILE_K) {
    __syncthreads();              // everyone done reading buffer ibuf^1
    ISSUE(kb + TILE_K, ibuf ^ 1);
    WAIT_ASYNC_6();               // oldest 6 (this step's tile) complete
    __syncthreads();              // buf[ibuf] fully populated by all waves
    compute(ibuf);
    ibuf ^= 1;
  }
  // final step
  WAIT_ASYNC_0();
  __syncthreads();
  compute(ibuf);
  #undef ISSUE

  // ---- epilogue ----
  const int colIn   = lane & 15;          // n within 16x16 tile
  const int rowHalf = (lane >> 4) * 8;    // C/D layout: M = v + 8*(lane/16)

  if (EPI == 0) {
    u16* o = (u16*)Out;
    #pragma unroll
    for (int ti = 0; ti < 2; ti++)
      #pragma unroll
      for (int tj = 0; tj < 4; tj++) {
        long long gcol = nBase + tj * 16 + colIn;
        #pragma unroll
        for (int v = 0; v < 8; v++) {
          int m = mBase + (wave * 2 + ti) * 16 + rowHalf + v;
          u16 val = f2bf(acc[ti][tj][v] + bias[m]);
          if (m < II) {                         // g rows
            o[(size_t)m * (size_t)ldo + (size_t)gcol] = val;
          } else if (m < 2 * II) {              // theta -> transposed aux
            aux[(size_t)gcol * II + (m - II)] = val;
          } else {                              // phi rows, compacted
            o[(size_t)(m - II) * (size_t)ldo + (size_t)gcol] = val;
          }
        }
      }
  } else if (EPI == 1) {
    u16* o = (u16*)Out + (long long)z * oBOff;
    #pragma unroll
    for (int ti = 0; ti < 2; ti++)
      #pragma unroll
      for (int tj = 0; tj < 4; tj++) {
        long long gcol = nBase + tj * 16 + colIn;
        #pragma unroll
        for (int v = 0; v < 8; v++) {
          int m = mBase + (wave * 2 + ti) * 16 + rowHalf + v;
          o[(size_t)gcol * (size_t)ldo + m] = f2bf(acc[ti][tj][v] * scale);
        }
      }
  } else {  // EPI == 3: fused bias + residual, fp32 out [B,C,H,W]
    float* o = (float*)Out;
    const int       bidx = (int)(nBase / NN);
    const long long nn0  = nBase - (long long)bidx * NN;
    #pragma unroll
    for (int ti = 0; ti < 2; ti++)
      #pragma unroll
      for (int tj = 0; tj < 4; tj++) {
        long long nn = nn0 + tj * 16 + colIn;
        #pragma unroll
        for (int v = 0; v < 8; v++) {
          int m = mBase + (wave * 2 + ti) * 16 + rowHalf + v;
          size_t addr = ((size_t)bidx * CC + (size_t)m) * NN + (size_t)nn;
          o[addr] = acc[ti][tj][v] + bias[m] + resid[addr];
        }
      }
  }
}

// ---------------- conversion kernels ----------------

// x [B,C,N] fp32 -> xbfT [B*N][C] bf16 via LDS-tiled 64x64 transpose.
__global__ __launch_bounds__(256) void conv_x_kernel(
    const float* __restrict__ x, u16* __restrict__ xbfT)
{
  __shared__ __align__(16) u16 sT[64 * 72];   // [n_local][c_local], padded

  const int t  = threadIdx.x;
  const int bx = blockIdx.x;   // n tile (49)
  const int by = blockIdx.y;   // c tile (8)
  const int bz = blockIdx.z;   // batch (16)

  // phase 1: read 16 n-contiguous floats per thread, convert, LDS transpose
  {
    int c_l = t >> 2;                 // 0..63
    int n16 = (t & 3) * 16;
    const float* src = x + ((size_t)(bz * CC + by * 64 + c_l) * NN
                            + (size_t)bx * 64 + n16);
    #pragma unroll
    for (int q = 0; q < 4; q++) {
      float4 f = *(const float4*)(src + q * 4);
      sT[(n16 + q * 4 + 0) * 72 + c_l] = f2bf(f.x);
      sT[(n16 + q * 4 + 1) * 72 + c_l] = f2bf(f.y);
      sT[(n16 + q * 4 + 2) * 72 + c_l] = f2bf(f.z);
      sT[(n16 + q * 4 + 3) * 72 + c_l] = f2bf(f.w);
    }
  }
  __syncthreads();
  // phase 2: write 16 c-contiguous bf16 per thread (2x b128), coalesced
  {
    int n_l = t >> 2;                 // 0..63
    int c16 = (t & 3) * 16;
    uint4 q0 = *(const uint4*)&sT[n_l * 72 + c16];
    uint4 q1 = *(const uint4*)&sT[n_l * 72 + c16 + 8];
    u16* dst = xbfT + ((size_t)(bz * NN + bx * 64 + n_l) * CC + by * 64 + c16);
    ((uint4*)dst)[0] = q0;
    ((uint4*)dst)[1] = q1;
  }
}

// weights -> bf16: Wcat [768][512] (g|theta|phi), Wfin [512][256], biascat[768]
__global__ __launch_bounds__(256) void conv_w_kernel(
    const float* __restrict__ g_w, const float* __restrict__ th_w,
    const float* __restrict__ ph_w, const float* __restrict__ g_b,
    const float* __restrict__ th_b, const float* __restrict__ ph_b,
    const float* __restrict__ W_w, u16* __restrict__ Wcat,
    u16* __restrict__ Wfin, float* __restrict__ biascat)
{
  int tid = blockIdx.x * 256 + threadIdx.x;  // 2048*256 = 524288
  if (tid < M3 * CC) {
    int m = tid >> 9, k = tid & 511;
    const float* src = (m < II) ? g_w : ((m < 2 * II) ? th_w : ph_w);
    int mm = m & (II - 1);
    Wcat[tid] = f2bf(src[mm * CC + k]);
  } else {
    int t2 = tid - M3 * CC;                  // 0..131071
    Wfin[t2] = f2bf(W_w[t2]);
  }
  if (tid < M3) {
    biascat[tid] = (tid < II) ? g_b[tid]
                 : (tid < 2 * II) ? th_b[tid - II] : ph_b[tid - 2 * II];
  }
}

// ---------------- workspace layout (bytes) ----------------
static const size_t OFF_XT  = 0;
static const size_t SZ_XT   = (size_t)NTOT * CC * 2;       // 51.4 MB
static const size_t OFF_P   = OFF_XT + SZ_XT;              // g|phi rows
static const size_t SZ_P    = (size_t)(2 * II) * NTOT * 2; // 51.4 MB
static const size_t OFF_THT = OFF_P + SZ_P;
static const size_t SZ_THT  = (size_t)NTOT * II * 2;       // 25.7 MB
static const size_t OFF_YT  = OFF_THT + SZ_THT;
static const size_t SZ_YT   = (size_t)NTOT * II * 2;       // 25.7 MB
static const size_t OFF_MT  = OFF_YT + SZ_YT;
static const size_t SZ_MT   = (size_t)BB * II * II * 2;    // 2 MB
static const size_t OFF_WC  = OFF_MT + SZ_MT;
static const size_t SZ_WC   = (size_t)M3 * CC * 2;
static const size_t OFF_WF  = OFF_WC + SZ_WC;
static const size_t SZ_WF   = (size_t)CC * II * 2;
static const size_t OFF_BC  = OFF_WF + SZ_WF;

extern "C" void kernel_launch(void* const* d_in, const int* in_sizes, int n_in,
                              void* d_out, int out_size, void* d_ws, size_t ws_size,
                              hipStream_t stream) {
  const float* x    = (const float*)d_in[0];
  const float* g_w  = (const float*)d_in[1];
  const float* g_b  = (const float*)d_in[2];
  const float* th_w = (const float*)d_in[3];
  const float* th_b = (const float*)d_in[4];
  const float* ph_w = (const float*)d_in[5];
  const float* ph_b = (const float*)d_in[6];
  const float* W_w  = (const float*)d_in[7];
  const float* W_b  = (const float*)d_in[8];

  char*  ws   = (char*)d_ws;
  u16*   xbfT = (u16*)(ws + OFF_XT);
  u16*   P    = (u16*)(ws + OFF_P);     // rows 0..255 g, 256..511 phi
  u16*   thT  = (u16*)(ws + OFF_THT);
  u16*   ybfT = (u16*)(ws + OFF_YT);
  u16*   Mt   = (u16*)(ws + OFF_MT);
  u16*   Wc   = (u16*)(ws + OFF_WC);
  u16*   Wf   = (u16*)(ws + OFF_WF);
  float* bc   = (float*)(ws + OFF_BC);

  conv_x_kernel<<<dim3(NN / 64, CC / 64, BB), 256, 0, stream>>>(x, xbfT);
  conv_w_kernel<<<(M3 * CC + CC * II) / 256, 256, 0, stream>>>(
      g_w, th_w, ph_w, g_b, th_b, ph_b, W_w, Wc, Wf, bc);

  // 1) projection: [g|theta|phi] = Wcat @ x + bias
  gemm_bf16_kernel<0, false>
      <<<dim3(NTOT / TILE_N, M3 / TILE_M, 1), 128, 0, stream>>>(
      Wc, CC, 0, xbfT, CC, 0, P, NTOT, 0, bc, nullptr, thT, 1.0f, CC);

  // 2) Mt[b][i][j] = (ph[b] @ g[b]^T)[j][i] / N   per batch, K = 3136
  gemm_bf16_kernel<1, false>
      <<<dim3(II / TILE_N, II / TILE_M, BB), 128, 0, stream>>>(
      P + (size_t)II * NTOT, NTOT, NN,
      P, NTOT, NN,
      Mt, II, (long long)II * II, nullptr, nullptr, nullptr,
      1.0f / (float)NN, NN);

  // 3) y^T: ybfT[col][i] = (Mt[b] @ th)[i][col]   (A block-diagonal, K = 256)
  gemm_bf16_kernel<1, true>
      <<<dim3(NTOT / TILE_N, II / TILE_M, 1), 128, 0, stream>>>(
      Mt, II, (long long)II * II,
      thT, II, 0,
      ybfT, II, 0, nullptr, nullptr, nullptr, 1.0f, II);

  // 4) out = x + Wfin @ y + W_b   (fp32, fused residual), K = 256
  gemm_bf16_kernel<3, false>
      <<<dim3(NTOT / TILE_N, CC / TILE_M, 1), 128, 0, stream>>>(
      Wf, II, 0, ybfT, II, 0, d_out, 0, 0, W_b, x, nullptr, 1.0f, II);
}
